// QuantLinear_62783831933672
// MI455X (gfx1250) — matrix operand order repaired and checked by reading, not compiled, not run on hardware
//
#include <hip/hip_runtime.h>
#include <hip/hip_bf16.h>
#include <stdint.h>

// Problem constants (from reference)
#define TOKENS 256
#define INF    4096
#define OUTF   11008
#define GS     128          // quantization group size
#define NGROUPS (INF / GS)  // 32
#define QZ_COLS (OUTF / 8)  // 1376 packed zero columns

typedef __attribute__((ext_vector_type(16))) _Float16 v16h;
typedef __attribute__((ext_vector_type(8)))  _Float16 v8h;
typedef __attribute__((ext_vector_type(2)))  _Float16 h2;
typedef __attribute__((ext_vector_type(8)))  float    v8f;

// ---------------------------------------------------------------------------
// Pass 1: convert activations f32 -> f16 into workspace (2 MB), writing each
// aligned 8-element K-chunk in permuted order [0,4,1,5,2,6,3,7].
// This matches the (j, j+4) nibble-pair order produced by the magic-number
// int4->f16 dequant on the B side, so the GEMM's A loads need no shuffling.
// 1,048,576 elements / 8 per thread -> 512 blocks x 256 threads.
// ---------------------------------------------------------------------------
__global__ void __launch_bounds__(256)
cvt_x_f16_perm_kernel(const float* __restrict__ x, _Float16* __restrict__ xh) {
    int i = (blockIdx.x * 256 + threadIdx.x) * 8;
    float4 a = *(const float4*)(x + i);
    float4 c = *(const float4*)(x + i + 4);
    v8h h;
    h[0] = (_Float16)a.x;  h[1] = (_Float16)c.x;   // K j paired with K j+4
    h[2] = (_Float16)a.y;  h[3] = (_Float16)c.y;
    h[4] = (_Float16)a.z;  h[5] = (_Float16)c.z;
    h[6] = (_Float16)a.w;  h[7] = (_Float16)c.w;
    *(v8h*)(xh + i) = h;
}

// ---------------------------------------------------------------------------
// Pass 2: W4A16 GEMM, int4 -> f16 magic-number dequant feeding
// v_wmma_f32_16x16x32_f16.
//
// Grid:  x = OUTF/128 = 86 column-tile blocks, y = TOKENS/64 = 4 row-tile blocks
// Block: 256 threads = 8 wave32; wave w owns columns [nbase + 16*w, +16).
// Each wave computes 4 M-tiles (64 rows): each dequantized B block feeds
// 4 WMMAs, and the pk_sub/pk_mul f16 dequant co-executes with the XDL.
// ---------------------------------------------------------------------------
__global__ void __launch_bounds__(256)
w4a16_wmma_kernel(const _Float16* __restrict__ xh,
                  const uint32_t* __restrict__ qweight,
                  const uint32_t* __restrict__ qzeros,
                  const float*    __restrict__ scales,
                  const float*    __restrict__ bias,
                  float*          __restrict__ out) {
    const int lane   = threadIdx.x & 31;
    const int wave   = threadIdx.x >> 5;          // 0..7
    const int lane16 = lane & 15;
    const int hi     = lane >> 4;                 // 0 or 1 (half-wave select)

    const int nbase = blockIdx.x * 128;           // column tile base
    const int mbase = blockIdx.y * 64;            // row tile base

    const int ncol = nbase + wave * 16 + lane16;  // this lane's output column
    const int kgrp = hi * 16;                     // B: lanes 16-31 hold K=16..31
    const int koff = hi * 16;                     // A: lanes 16-31 hold K=16..31

    const int zshift = 4 * (ncol & 7);
    const int zcol   = ncol >> 3;

    v8f acc[4];
#pragma unroll
    for (int mt = 0; mt < 4; ++mt) acc[mt] = (v8f){0.f,0.f,0.f,0.f,0.f,0.f,0.f,0.f};

    const _Float16* aptr[4];
#pragma unroll
    for (int mt = 0; mt < 4; ++mt)
        aptr[mt] = xh + (size_t)(mbase + mt * 16 + lane16) * INF + koff;

    const uint32_t* qwp = qweight + ncol;         // column base, stride OUTF per row

    for (int g = 0; g < NGROUPS; ++g) {
        // Per-group dequant constants for this lane's column.
        const uint32_t qz = qzeros[g * QZ_COLS + zcol];
        const float scalef = scales[g * OUTF + ncol];
        const int   z      = (int)((qz >> zshift) & 0xF) + 1;
        const _Float16 hs  = (_Float16)scalef;
        const _Float16 hzv = (_Float16)(float)(1024 + z);  // exact in f16
        const h2 hs2  = (h2){hs, hs};
        const h2 hz2  = (h2){hzv, hzv};

        // Prefetch next group's first packed rows (global_prefetch_b8)
        if (g + 1 < NGROUPS) {
            __builtin_prefetch(qwp + (size_t)((g + 1) * (GS / 8)) * OUTF, 0, 1);
        }

#pragma unroll
        for (int kk = 0; kk < 4; ++kk) {
            const int kb = g * GS + kk * 32;      // K-step base (32 per WMMA)

            // ---- dequantize B: 16 K-values of column `ncol` ----
            const int krow = (kb + kgrp) >> 3;    // packed row (8 nibbles each)
            const uint32_t q0 = qwp[(size_t)krow * OUTF];
            const uint32_t q1 = qwp[(size_t)(krow + 1) * OUTF];

            union { v16h v; h2 h[8]; uint32_t u[8]; } b;
#pragma unroll
            for (int j = 0; j < 4; ++j) {
                // halves = f16(1024 + nib_j), f16(1024 + nib_{j+4})
                b.u[j]     = ((q0 >> (4 * j)) & 0x000F000Fu) | 0x64006400u;
                b.u[4 + j] = ((q1 >> (4 * j)) & 0x000F000Fu) | 0x64006400u;
            }
#pragma unroll
            for (int j = 0; j < 8; ++j) {
                // (h - (1024+z)) is exact (Sterbenz); then scale.
                b.h[j] = (b.h[j] - hz2) * hs2;    // v_pk_sub_f16 + v_pk_mul_f16
            }

            // ---- 4 M-tiles share this B block ----
#pragma unroll
            for (int mt = 0; mt < 4; ++mt) {
                const _Float16* pa = aptr[mt] + kb;
                v8h a0 = *(const v8h*)(pa);       // permuted K = kb+koff chunk
                v8h a1 = *(const v8h*)(pa + 8);   // permuted K = kb+8+koff chunk
                v16h a;
#pragma unroll
                for (int j = 0; j < 8; ++j) { a[j] = a0[j]; a[8 + j] = a1[j]; }

                acc[mt] = __builtin_amdgcn_wmma_f32_16x16x32_f16(
                    /*neg_a=*/false, a, /*neg_b=*/false, b.v,
                    /*c_mod=*/(short)0, acc[mt],
                    /*reuse_a=*/false, /*reuse_b=*/false);
            }
        }
    }

    // Store: C/D layout -> VGPR i holds M = i + 8*hi, N = lane16
    const float bv = bias[ncol];
#pragma unroll
    for (int mt = 0; mt < 4; ++mt) {
        const int rowb = mbase + mt * 16 + hi * 8;
#pragma unroll
        for (int i = 0; i < 8; ++i) {
            out[(size_t)(rowb + i) * OUTF + ncol] = acc[mt][i] + bv;
        }
    }
}

// ---------------------------------------------------------------------------
// Launcher
// ---------------------------------------------------------------------------
extern "C" void kernel_launch(void* const* d_in, const int* in_sizes, int n_in,
                              void* d_out, int out_size, void* d_ws, size_t ws_size,
                              hipStream_t stream) {
    const float*    x       = (const float*)   d_in[0];
    const uint32_t* qweight = (const uint32_t*)d_in[1];
    const uint32_t* qzeros  = (const uint32_t*)d_in[2];
    const float*    scales  = (const float*)   d_in[3];
    const float*    bias    = (const float*)   d_in[4];
    float*          out     = (float*)d_out;

    _Float16* xh = (_Float16*)d_ws;   // 256*4096*2 = 2 MB of workspace

    // Pass 1: f32 -> f16 activations (K-permuted per 8-chunk)
    cvt_x_f16_perm_kernel<<<(TOKENS * INF) / (256 * 8), 256, 0, stream>>>(x, xh);

    // Pass 2: WMMA GEMM
    dim3 grid(OUTF / 128, TOKENS / 64);
    w4a16_wmma_kernel<<<grid, 256, 0, stream>>>(xh, qweight, qzeros, scales, bias, out);
}
